// ZZKNet_svt_86294482911322
// MI455X (gfx1250) — matrix operand order repaired and checked by reading, not compiled, not run on hardware
//
#include <hip/hip_runtime.h>
#include <hip/hip_bf16.h>

// ---------------------------------------------------------------------------
// DTCWT + complex block-MLP (SVT) for MI455X (gfx1250, wave32).
//
// Pipeline (all intermediates in d_ws, channel-minor layout, fits 192MB L2):
//   k_pack_w   : pack W1/W2 (with complex signs folded in) into per-lane
//                WMMA A-matrix register layout, f16.
//   k_fwd_rows : x (NCHW) --H0/H1 row filters--> lo,hi (NHWC f32)
//   k_fwd_cols : col filters + q2c decimation -> yl (NHWC) + 6 bands f16
//   k_svt_wmma : 2-layer complex MLP per 16-ch block via v_wmma_f32_16x16x32_f16
//                (K=32 packs [xr|xi] so one WMMA = full complex MAC)
//   k_inv_c2q  : bands -> LH/HL/HH full-res (NHWC)
//   k_inv_cols : G0/G1 column synthesis -> lo2,hi2
//   k_inv_rows : G0/G1 row synthesis -> d_out (NCHW)
//
// Workspace requirement: ~192.7 MB (with buffer reuse).
// ---------------------------------------------------------------------------

typedef __attribute__((ext_vector_type(16))) _Float16 v16h;
typedef __attribute__((ext_vector_type(8)))  float    v8f;

#define ISQ2 0.70710678118654752440f

__constant__ float c_H0[13] = {
  -0.0017578125f, 0.0f, 0.022265625f, -0.046875f, -0.0482421875f, 0.296875f,
  0.55546875f, 0.296875f, -0.0482421875f, -0.046875f, 0.022265625f, 0.0f,
  -0.0017578125f };
__constant__ float c_H1[19] = {
  -7.06261e-05f, 0.0f, 0.0013419f, -0.0018834f, -0.0071568f, 0.023856f,
  0.0556431f, -0.0516881f, -0.2997576f, 0.5594308f, -0.2997576f, -0.0516881f,
  0.0556431f, 0.023856f, -0.0071568f, -0.0018834f, 0.0013419f, 0.0f,
  -7.06261e-05f };
// G0 = H1 * (+1 odd idx, -1 even idx)
__constant__ float c_G0[19] = {
  7.06261e-05f, 0.0f, -0.0013419f, -0.0018834f, 0.0071568f, 0.023856f,
  -0.0556431f, -0.0516881f, 0.2997576f, 0.5594308f, 0.2997576f, -0.0516881f,
  -0.0556431f, 0.023856f, 0.0071568f, -0.0018834f, -0.0013419f, 0.0f,
  7.06261e-05f };
// G1 = H0 * (-1 even idx, +1 odd idx)
__constant__ float c_G1[13] = {
  0.0017578125f, 0.0f, -0.022265625f, -0.046875f, 0.0482421875f, 0.296875f,
  -0.55546875f, 0.296875f, 0.0482421875f, -0.046875f, -0.022265625f, 0.0f,
  0.0017578125f };

__device__ __forceinline__ int symref(int m, int N) {
  if (m < 0)  m = -1 - m;
  if (m >= N) m = 2 * N - 1 - m;
  return m;
}

// --------------------------------------------------------------------------
// Weight packer: A[mat][blk] is the 16x32 matrix  [k, d] with
//   r-output (mat even): [ W_r[d,k] | -W_i[d-16,k] ]
//   i-output (mat odd):  [ W_i[d,k] |  W_r[d-16,k] ]
// packed per lane as the f16 16x32 A operand in this file's contiguous k order (same as B):
//   lane<16:  element e -> K = e       (K in {0..15}),  M=lane
//   lane>=16: element e -> K = 16+e    (K in {16..31}), M=lane-16
// --------------------------------------------------------------------------
__global__ __launch_bounds__(256) void k_pack_w(const float* __restrict__ w1,
                                                const float* __restrict__ w2,
                                                _Float16* __restrict__ wp) {
  int idx  = blockIdx.x * 256 + threadIdx.x;   // 4 mats * 4 blks * 32 lanes * 16 = 8192
  int e    = idx & 15;
  int lane = (idx >> 4) & 31;
  int blk  = (idx >> 9) & 3;
  int mat  = idx >> 11;                        // 0=A1r 1=A1i 2=A2r 3=A2i
  int M    = lane & 15;
  int K    = (lane < 16) ? e : 16 + e;
  const float* w = (mat < 2) ? w1 : w2;        // (2, 4, 16, 16)
  int d = (K < 16) ? K : K - 16;
  float wr = w[blk * 256 + d * 16 + M];
  float wi = w[1024 + blk * 256 + d * 16 + M];
  float val = ((mat & 1) == 0) ? ((K < 16) ? wr : -wi)
                               : ((K < 16) ? wi :  wr);
  wp[idx] = (_Float16)val;
}

// --------------------------------------------------------------------------
// Forward row filters: x (NCHW) -> lo = x*H0, hi = x*H1 along W (NHWC out).
// --------------------------------------------------------------------------
__global__ __launch_bounds__(256) void k_fwd_rows(const float* __restrict__ x,
                                                  float* __restrict__ lo,
                                                  float* __restrict__ hi) {
  int i = blockIdx.x * 256 + threadIdx.x;      // 32*64*56*56, NCHW order
  int w = i % 56;  int t1 = i / 56;
  int h = t1 % 56; int t2 = t1 / 56;
  int c = t2 % 64; int b  = t2 / 64;
  const float* xr = x + (((size_t)(b * 64 + c)) * 56 + h) * 56;
  float slo = 0.f, shi = 0.f;
#pragma unroll
  for (int t = 0; t < 19; ++t) {
    float v = xr[symref(w + t - 9, 56)];
    shi += c_H1[t] * v;
    if (t >= 3 && t < 16) slo += c_H0[t - 3] * v;
  }
  size_t o = (((size_t)b * 56 + h) * 56 + w) * 64 + c;
  lo[o] = slo;
  hi[o] = shi;
}

// --------------------------------------------------------------------------
// Forward column filters + q2c: lo,hi (NHWC) -> yl (NHWC) + 6 f16 bands
// bands layout: (band, B, 28, 28, C) channel-minor (WMMA-friendly).
// --------------------------------------------------------------------------
__global__ __launch_bounds__(256) void k_fwd_cols(const float* __restrict__ lo,
                                                  const float* __restrict__ hi,
                                                  float* __restrict__ yl,
                                                  _Float16* __restrict__ bRh,
                                                  _Float16* __restrict__ bIh) {
  int i  = blockIdx.x * 256 + threadIdx.x;     // 32*28*28*64, c fastest
  int c  = i & 63;  int t1 = i >> 6;
  int w2 = t1 % 28; int t2 = t1 / 28;
  int h2 = t2 % 28; int b  = t2 / 28;

  float qlh[4], qhl[4], qhh[4];
#pragma unroll
  for (int dh = 0; dh < 2; ++dh) {
    int hb = 2 * h2 + dh;
#pragma unroll
    for (int dw = 0; dw < 2; ++dw) {
      int w = 2 * w2 + dw;
      float sll = 0.f, slh = 0.f, shl = 0.f, shh = 0.f;
#pragma unroll
      for (int t = 0; t < 19; ++t) {
        int r = symref(hb + t - 9, 56);
        size_t a = (((size_t)b * 56 + r) * 56 + w) * 64 + c;
        float lv = lo[a], hv = hi[a];
        slh += c_H1[t] * lv;
        shh += c_H1[t] * hv;
        if (t >= 3 && t < 16) { float f = c_H0[t - 3]; sll += f * lv; shl += f * hv; }
      }
      yl[(((size_t)b * 56 + hb) * 56 + w) * 64 + c] = sll;
      int q = dh * 2 + dw;
      qlh[q] = slh * ISQ2; qhl[q] = shl * ISQ2; qhh[q] = shh * ISQ2;
    }
  }
  size_t base    = (((size_t)b * 28 + h2) * 28 + w2) * 64 + c;
  size_t bstride = (size_t)32 * 28 * 28 * 64;
  auto store = [&](int band, float r, float im) {
    bRh[base + (size_t)band * bstride] = (_Float16)r;
    bIh[base + (size_t)band * bstride] = (_Float16)im;
  };
  // q2c: a=q0, b=q1, c=q2, d=q3 ; first=(a-d, b+c), second=(a+d, b-c)
  store(0, qlh[0] - qlh[3], qlh[1] + qlh[2]);
  store(5, qlh[0] + qlh[3], qlh[1] - qlh[2]);
  store(1, qhh[0] - qhh[3], qhh[1] + qhh[2]);
  store(4, qhh[0] + qhh[3], qhh[1] - qhh[2]);
  store(2, qhl[0] - qhl[3], qhl[1] + qhl[2]);
  store(3, qhl[0] + qhl[3], qhl[1] - qhl[2]);
}

// --------------------------------------------------------------------------
// SVT core: per wave, one (16-position tile x 16-channel block).
// D[k,s] = A(weights 16x32) x B(data 32x16), K packs [real(16) | imag(16)],
// so ONE v_wmma_f32_16x16x32_f16 performs a full complex MAC.
// Layer1 -> bias+ReLU -> half-wave shfl_xor swap builds layer2's B -> Layer2.
// --------------------------------------------------------------------------
__global__ __launch_bounds__(256) void k_svt_wmma(const _Float16* __restrict__ bRh,
                                                  const _Float16* __restrict__ bIh,
                                                  const _Float16* __restrict__ wp,
                                                  const float* __restrict__ b1,
                                                  const float* __restrict__ b2,
                                                  float* __restrict__ oR,
                                                  float* __restrict__ oI) {
  int unit = blockIdx.x * 8 + (threadIdx.x >> 5);   // 37632 waves total, exact
  int lane = threadIdx.x & 31;
  int blk  = unit & 3;
  int tile = unit >> 2;
  int srow = lane & 15;
  bool isHi = lane >= 16;
  int base  = isHi ? 8 : 0;
  size_t p  = (size_t)tile * 16 + srow;             // position (band,b,h2,w2 flat)

  // B operand, layer 1: lanes<16 carry K=0..15 (=xr block), lanes>=16 K=16..31 (=xi)
  const _Float16* dsrc = isHi ? bIh : bRh;
  v16h Bdat = *(const v16h*)(dsrc + p * 64 + blk * 16);

  // A operands: pre-packed per-lane weights
  const _Float16* wl = wp + ((size_t)blk * 32 + lane) * 16;
  const size_t MS = (size_t)4 * 32 * 16;
  v16h A1r = *(const v16h*)(wl);
  v16h A1i = *(const v16h*)(wl + MS);
  v16h A2r = *(const v16h*)(wl + 2 * MS);
  v16h A2i = *(const v16h*)(wl + 3 * MS);

  v8f z = {};
  v8f accR = __builtin_amdgcn_wmma_f32_16x16x32_f16(false, A1r, false, Bdat, (short)0, z, false, false);
  v8f accI = __builtin_amdgcn_wmma_f32_16x16x32_f16(false, A1i, false, Bdat, (short)0, z, false, false);

  v8f bR1 = *(const v8f*)(b1 + blk * 16 + base);
  v8f bI1 = *(const v8f*)(b1 + 64 + blk * 16 + base);
#pragma unroll
  for (int v = 0; v < 8; ++v) {
    accR[v] = fmaxf(accR[v] + bR1[v], 0.0f);
    accI[v] = fmaxf(accI[v] + bI1[v], 0.0f);
  }

  // Build layer-2 B operand: D-layout (lane = spatial col, rows = channels)
  // -> B-layout (lane<16 needs r1 ch0..15, lane>=16 needs i1 ch0..15) via xor-16 swap.
  v16h B2;
#pragma unroll
  for (int cc = 0; cc < 8; ++cc) {
    float r_own = accR[cc];
    float i_own = accI[cc];
    float r_x = __shfl_xor(r_own, 16, 32);
    float i_x = __shfl_xor(i_own, 16, 32);
    B2[cc]     = (_Float16)(isHi ? i_x  : r_own);  // channel cc
    B2[cc + 8] = (_Float16)(isHi ? i_own : r_x);   // channel cc+8
  }

  v8f accR2 = __builtin_amdgcn_wmma_f32_16x16x32_f16(false, A2r, false, B2, (short)0, z, false, false);
  v8f accI2 = __builtin_amdgcn_wmma_f32_16x16x32_f16(false, A2i, false, B2, (short)0, z, false, false);

  v8f bR2 = *(const v8f*)(b2 + blk * 16 + base);
  v8f bI2 = *(const v8f*)(b2 + 64 + blk * 16 + base);
#pragma unroll
  for (int v = 0; v < 8; ++v) { accR2[v] += bR2[v]; accI2[v] += bI2[v]; }

  size_t off = p * 64 + (size_t)blk * 16 + base;    // contiguous 8 floats / lane
  *(v8f*)(oR + off) = accR2;
  *(v8f*)(oI + off) = accI2;
}

// --------------------------------------------------------------------------
// Inverse c2q: band pairs (0,5)->LH, (1,4)->HH, (2,3)->HL, full-res NHWC.
// --------------------------------------------------------------------------
__global__ __launch_bounds__(256) void k_inv_c2q(const float* __restrict__ oR,
                                                 const float* __restrict__ oI,
                                                 float* __restrict__ LH,
                                                 float* __restrict__ HL,
                                                 float* __restrict__ HH) {
  int i  = blockIdx.x * 256 + threadIdx.x;     // 32*28*28*64, c fastest
  int c  = i & 63;  int t1 = i >> 6;
  int w2 = t1 % 28; int t2 = t1 / 28;
  int h2 = t2 % 28; int b  = t2 / 28;
  size_t bstride = (size_t)32 * 28 * 28 * 64;
  size_t base    = (((size_t)b * 28 + h2) * 28 + w2) * 64 + c;
  size_t o00     = (((size_t)b * 56 + 2 * h2) * 56 + 2 * w2) * 64 + c;
  auto c2q = [&](int ba, int bb, float* dst) {
    float z1r = oR[base + (size_t)ba * bstride], z1i = oI[base + (size_t)ba * bstride];
    float z2r = oR[base + (size_t)bb * bstride], z2i = oI[base + (size_t)bb * bstride];
    dst[o00]               = (z1r + z2r) * ISQ2;
    dst[o00 + 64]          = (z1i + z2i) * ISQ2;
    dst[o00 + 56 * 64]     = (z1i - z2i) * ISQ2;
    dst[o00 + 56 * 64 + 64] = (z2r - z1r) * ISQ2;
  };
  c2q(0, 5, LH);
  c2q(1, 4, HH);
  c2q(2, 3, HL);
}

// --------------------------------------------------------------------------
// Inverse column synthesis: lo2 = yl*G0 + LH*G1 ; hi2 = HL*G0 + HH*G1 (axis H)
// --------------------------------------------------------------------------
__global__ __launch_bounds__(256) void k_inv_cols(const float* __restrict__ yl,
                                                  const float* __restrict__ LH,
                                                  const float* __restrict__ HL,
                                                  const float* __restrict__ HH,
                                                  float* __restrict__ lo2,
                                                  float* __restrict__ hi2) {
  int i = blockIdx.x * 256 + threadIdx.x;      // 32*56*56*64, c fastest
  int c = i & 63;  int t1 = i >> 6;
  int w = t1 % 56; int t2 = t1 / 56;
  int h = t2 % 56; int b  = t2 / 56;
  float slo = 0.f, shi = 0.f;
#pragma unroll
  for (int t = 0; t < 19; ++t) {
    int r = symref(h + t - 9, 56);
    size_t a = (((size_t)b * 56 + r) * 56 + w) * 64 + c;
    slo += c_G0[t] * yl[a];
    shi += c_G0[t] * HL[a];
    if (t >= 3 && t < 16) { float g = c_G1[t - 3]; slo += g * LH[a]; shi += g * HH[a]; }
  }
  size_t o = (((size_t)b * 56 + h) * 56 + w) * 64 + c;
  lo2[o] = slo;
  hi2[o] = shi;
}

// --------------------------------------------------------------------------
// Inverse row synthesis + NHWC -> NCHW: out = lo2*G0 + hi2*G1 along W.
// --------------------------------------------------------------------------
__global__ __launch_bounds__(256) void k_inv_rows(const float* __restrict__ lo2,
                                                  const float* __restrict__ hi2,
                                                  float* __restrict__ out) {
  int i = blockIdx.x * 256 + threadIdx.x;      // NCHW order (w fastest) -> coalesced out
  int w = i % 56;  int t1 = i / 56;
  int h = t1 % 56; int t2 = t1 / 56;
  int c = t2 % 64; int b  = t2 / 64;
  float s = 0.f;
#pragma unroll
  for (int t = 0; t < 19; ++t) {
    int m = symref(w + t - 9, 56);
    size_t a = (((size_t)b * 56 + h) * 56 + m) * 64 + c;
    s += c_G0[t] * lo2[a];
    if (t >= 3 && t < 16) s += c_G1[t - 3] * hi2[a];
  }
  out[i] = s;
}

// ---------------------------------------------------------------------------

extern "C" void kernel_launch(void* const* d_in, const int* in_sizes, int n_in,
                              void* d_out, int out_size, void* d_ws, size_t ws_size,
                              hipStream_t stream) {
  (void)in_sizes; (void)n_in; (void)out_size; (void)ws_size;
  const float* x  = (const float*)d_in[0];
  const float* w1 = (const float*)d_in[1];
  const float* w2 = (const float*)d_in[2];
  const float* b1 = (const float*)d_in[3];
  const float* b2 = (const float*)d_in[4];
  float* out = (float*)d_out;
  char*  ws  = (char*)d_ws;

  const size_t S_PLANE = (size_t)32 * 56 * 56 * 64 * 4;   // 25,690,112 B
  const size_t S_BANDH = (size_t)6 * 32 * 28 * 28 * 64 * 2; // 19,267,584 B
  const size_t S_BANDF = (size_t)6 * 32 * 28 * 28 * 64 * 4; // 38,535,168 B

  float*    lo  = (float*)(ws);
  float*    hi  = (float*)(ws + S_PLANE);
  float*    yl  = (float*)(ws + 2 * S_PLANE);
  _Float16* bRh = (_Float16*)(ws + 3 * S_PLANE);
  _Float16* bIh = (_Float16*)(ws + 3 * S_PLANE + S_BANDH);
  float*    oR  = (float*)(ws + 3 * S_PLANE + 2 * S_BANDH);
  float*    oI  = (float*)(ws + 3 * S_PLANE + 2 * S_BANDH + S_BANDF);
  _Float16* wp  = (_Float16*)(ws + 3 * S_PLANE + 2 * S_BANDH + 2 * S_BANDF);
  // Lifetime-based reuse (dead buffers):
  float* LH  = (float*)(ws + 3 * S_PLANE);                    // over bRh/bIh
  float* HH  = lo;                                            // over lo
  float* HL  = hi;                                            // over hi
  float* lo2 = oR;                                            // over oR head
  float* hi2 = (float*)((char*)oR + S_PLANE);                 // over oR tail/oI head

  k_pack_w  <<<32,    256, 0, stream>>>(w1, w2, wp);
  k_fwd_rows<<<25088, 256, 0, stream>>>(x, lo, hi);
  k_fwd_cols<<<6272,  256, 0, stream>>>(lo, hi, yl, bRh, bIh);
  k_svt_wmma<<<4704,  256, 0, stream>>>(bRh, bIh, wp, b1, b2, oR, oI);
  k_inv_c2q <<<6272,  256, 0, stream>>>(oR, oI, LH, HL, HH);
  k_inv_cols<<<25088, 256, 0, stream>>>(yl, LH, HL, HH, lo2, hi2);
  k_inv_rows<<<25088, 256, 0, stream>>>(lo2, hi2, out);
}
